// Smart_tiny_60773787238570
// MI455X (gfx1250) — matrix operand order repaired and checked by reading, not compiled, not run on hardware
//
#include <hip/hip_runtime.h>

#define E_EDGES 2048
#define N_NODES 512
#define NEG_INF (-3.402823466e38f)

typedef __attribute__((ext_vector_type(16))) _Float16 v16h;
typedef __attribute__((ext_vector_type(8)))  float    v8f;

// ---------------------------------------------------------------------------
// Deterministic block reduction: wave32 shuffle -> LDS across waves -> direct
// store of this block's partial sums (no float atomics).
// ---------------------------------------------------------------------------
template<int NV>
__device__ __forceinline__ void block_reduce_store(const float* v, float* dst, float* lds)
{
    const int lane = threadIdx.x & 31;
    const int wv   = threadIdx.x >> 5;
#pragma unroll
    for (int k = 0; k < NV; ++k) {
        float s = v[k];
#pragma unroll
        for (int off = 16; off > 0; off >>= 1) s += __shfl_down(s, off, 32);
        if (lane == 0) lds[wv * NV + k] = s;
    }
    __syncthreads();
    if ((int)threadIdx.x < NV) {
        float s = 0.f;
        const int nw = blockDim.x >> 5;
        for (int i = 0; i < nw; ++i) s += lds[i * NV + threadIdx.x];
        dst[threadIdx.x] = s;
    }
    __syncthreads();
}

// ---------------------------------------------------------------------------
// Node multiplicity histogram (integer atomics: deterministic).
// ---------------------------------------------------------------------------
__global__ void count_kernel(const int* __restrict__ ei, int* __restrict__ c0,
                             int* __restrict__ c1, int E)
{
    int t = blockIdx.x * blockDim.x + threadIdx.x;
    if (t < E) {
        atomicAdd(&c0[ei[t]], 1);
        atomicAdd(&c1[ei[E + t]], 1);
    }
}

// ---------------------------------------------------------------------------
// BN affine from deterministic partial sums: aff[c]=scale, aff[C+c]=shift.
// ---------------------------------------------------------------------------
__global__ void bn_affine_kernel(const float* __restrict__ partials, int nblocks, int stride,
                                 int offS, int offQ, int C, const float* __restrict__ g,
                                 const float* __restrict__ beta, float invT,
                                 float* __restrict__ aff)
{
    int c = threadIdx.x;
    if (c >= C) return;
    float s = 0.f, q = 0.f;
    for (int b = 0; b < nblocks; ++b) {
        s += partials[b * stride + offS + c];
        q += partials[b * stride + offQ + c];
    }
    float mean = s * invT;
    float var  = q * invT - mean * mean;
    float sc   = g[c] / sqrtf(var + 1e-5f);
    aff[c]     = sc;
    aff[C + c] = beta[c] - mean * sc;
}

// ---------------------------------------------------------------------------
// Stage 1 (per NODE, both conv1 and conv1b fused): raw x -> conv3x3 (1->4)x2,
// pooled (k2 s2 p1) pre-BN max stored; count-weighted sum/sumsq partials.
// partial layout per block: [sA0..3, qA0..3, sB0..3, qB0..3]
// ---------------------------------------------------------------------------
__global__ void conv1_kernel(const float* __restrict__ x,
                             const int* __restrict__ cnt0, const int* __restrict__ cnt1,
                             const float* __restrict__ wA, const float* __restrict__ bA,
                             const float* __restrict__ wB, const float* __restrict__ bB,
                             float* __restrict__ PA, float* __restrict__ PB,
                             float* __restrict__ partials)
{
    __shared__ float lds[8 * 16];
    const int IH = 320, IW = 100, OH = 161, OW = 51;
    const int total = N_NODES * OH * OW;
    int tid = blockIdx.x * blockDim.x + threadIdx.x;

    float part[16];
#pragma unroll
    for (int k = 0; k < 16; ++k) part[k] = 0.f;

    if (tid < total) {
        int n   = tid / (OH * OW);
        int rem = tid % (OH * OW);
        int oi  = rem / OW, oj = rem % OW;
        int r0  = 2 * oi - 1, c0 = 2 * oj - 1;
        const float* xin = x + (long)n * IH * IW;

        // 4x4 input patch covering the 2x2 window of 3x3 conv pixels
        float p[4][4];
#pragma unroll
        for (int a = 0; a < 4; ++a) {
            int r = r0 - 1 + a;
#pragma unroll
            for (int b = 0; b < 4; ++b) {
                int c = c0 - 1 + b;
                p[a][b] = (r >= 0 && r < IH && c >= 0 && c < IW) ? xin[r * IW + c] : 0.f;
            }
        }

        float mA[4], sA[4], qA[4], mB[4], sB[4], qB[4];
#pragma unroll
        for (int ch = 0; ch < 4; ++ch) {
            mA[ch] = NEG_INF; mB[ch] = NEG_INF;
            sA[ch] = qA[ch] = sB[ch] = qB[ch] = 0.f;
        }

#pragma unroll
        for (int a0 = 0; a0 < 2; ++a0) {
            int r = r0 + a0;
            if (r < 0 || r >= IH) continue;
#pragma unroll
            for (int b0 = 0; b0 < 2; ++b0) {
                int c = c0 + b0;
                if (c < 0 || c >= IW) continue;
#pragma unroll
                for (int ch = 0; ch < 4; ++ch) {
                    float va = bA[ch], vb = bB[ch];
#pragma unroll
                    for (int i = 0; i < 3; ++i)
#pragma unroll
                        for (int j = 0; j < 3; ++j) {
                            float xv = p[a0 + i][b0 + j];
                            va += xv * wA[ch * 9 + i * 3 + j];
                            vb += xv * wB[ch * 9 + i * 3 + j];
                        }
                    mA[ch] = fmaxf(mA[ch], va); sA[ch] += va; qA[ch] += va * va;
                    mB[ch] = fmaxf(mB[ch], vb); sB[ch] += vb; qB[ch] += vb * vb;
                }
            }
        }

        float wa = (float)cnt0[n], wb = (float)cnt1[n];
#pragma unroll
        for (int ch = 0; ch < 4; ++ch) {
            PA[((long)n * 4 + ch) * (OH * OW) + rem] = mA[ch];
            PB[((long)n * 4 + ch) * (OH * OW) + rem] = mB[ch];
            part[ch]      = wa * sA[ch];
            part[4 + ch]  = wa * qA[ch];
            part[8 + ch]  = wb * sB[ch];
            part[12 + ch] = wb * qB[ch];
        }
    }
    block_reduce_store<16>(part, partials + (long)blockIdx.x * 16, lds);
}

// ---------------------------------------------------------------------------
// Generic stage: input = relu(aff * pooledPreBN + shift) * sigmoid(seb2[ci]),
// zero-padded conv3x3 CIN->COUT, pooled (PKxPK stride PS pad 1) pre-BN max.
// cnt != null -> count-weighted statistics (per-node stages).
// ---------------------------------------------------------------------------
template<int CIN, int COUT, int PK, int PS>
__global__ void conv_stage_kernel(const float* __restrict__ inP, const float* __restrict__ aff,
                                  const float* __restrict__ seb, const float* __restrict__ wgt,
                                  const float* __restrict__ bias, const int* __restrict__ cnt,
                                  float* __restrict__ outP, float* __restrict__ partials,
                                  int N, int IH, int IW, int OH, int OW)
{
    __shared__ float lds[8 * 16];
    const int NV = 2 * COUT;
    const int total = N * OH * OW;
    int tid = blockIdx.x * blockDim.x + threadIdx.x;

    float part[NV];
#pragma unroll
    for (int k = 0; k < NV; ++k) part[k] = 0.f;

    if (tid < total) {
        int n   = tid / (OH * OW);
        int rem = tid % (OH * OW);
        int oi  = rem / OW, oj = rem % OW;

        float a_s[CIN], a_b[CIN], se[CIN], bi[COUT];
#pragma unroll
        for (int ci = 0; ci < CIN; ++ci) {
            a_s[ci] = aff ? aff[ci]       : 1.f;
            a_b[ci] = aff ? aff[CIN + ci] : 0.f;
            se[ci]  = seb ? 1.f / (1.f + expf(-seb[ci])) : 1.f;
        }
#pragma unroll
        for (int co = 0; co < COUT; ++co) bi[co] = bias[co];

        float m[COUT], s[COUT], q[COUT];
#pragma unroll
        for (int co = 0; co < COUT; ++co) { m[co] = NEG_INF; s[co] = 0.f; q[co] = 0.f; }

        int r0 = PS * oi - 1, c0 = PS * oj - 1;
#pragma unroll
        for (int a0 = 0; a0 < PK; ++a0) {
            int r = r0 + a0;
            if (r < 0 || r >= IH) continue;
#pragma unroll
            for (int b0 = 0; b0 < PK; ++b0) {
                int c = c0 + b0;
                if (c < 0 || c >= IW) continue;
                float acc[COUT];
#pragma unroll
                for (int co = 0; co < COUT; ++co) acc[co] = bi[co];
#pragma unroll
                for (int ci = 0; ci < CIN; ++ci) {
                    const float* ip = inP + ((long)n * CIN + ci) * IH * IW;
#pragma unroll
                    for (int di = 0; di < 3; ++di) {
                        int rr = r + di - 1;
                        if (rr < 0 || rr >= IH) continue;
#pragma unroll
                        for (int dj = 0; dj < 3; ++dj) {
                            int cc = c + dj - 1;
                            if (cc < 0 || cc >= IW) continue;
                            float raw = ip[rr * IW + cc];
                            float v = fmaxf(0.f, a_s[ci] * raw + a_b[ci]) * se[ci];
#pragma unroll
                            for (int co = 0; co < COUT; ++co)
                                acc[co] += v * wgt[(co * CIN + ci) * 9 + di * 3 + dj];
                        }
                    }
                }
#pragma unroll
                for (int co = 0; co < COUT; ++co) {
                    m[co] = fmaxf(m[co], acc[co]);
                    s[co] += acc[co];
                    q[co] += acc[co] * acc[co];
                }
            }
        }
        float wN = cnt ? (float)cnt[n] : 1.f;
#pragma unroll
        for (int co = 0; co < COUT; ++co) {
            outP[((long)n * COUT + co) * (OH * OW) + rem] = m[co];
            part[co]        = wN * s[co];
            part[COUT + co] = wN * q[co];
        }
    }
    block_reduce_store<NV>(part, partials + (long)blockIdx.x * NV, lds);
}

// ---------------------------------------------------------------------------
// Stage 3 (per EDGE): input = relu(affA(P2a[src])) - relu(affB(P2b[dst])),
// conv3x3 4->4, pool k2 s2 p1.
// ---------------------------------------------------------------------------
__global__ void conv3_diff_kernel(const float* __restrict__ PA, const float* __restrict__ PB,
                                  const float* __restrict__ affA, const float* __restrict__ affB,
                                  const int* __restrict__ src, const int* __restrict__ dst,
                                  const float* __restrict__ wgt, const float* __restrict__ bias,
                                  float* __restrict__ outP, float* __restrict__ partials,
                                  int N, int IH, int IW, int OH, int OW)
{
    __shared__ float lds[8 * 16];
    const int total = N * OH * OW;
    int tid = blockIdx.x * blockDim.x + threadIdx.x;

    float part[8];
#pragma unroll
    for (int k = 0; k < 8; ++k) part[k] = 0.f;

    if (tid < total) {
        int n   = tid / (OH * OW);
        int rem = tid % (OH * OW);
        int oi  = rem / OW, oj = rem % OW;
        int na = src[n], nb = dst[n];

        float aS[4], aB[4], bS[4], bB[4], bi[4];
#pragma unroll
        for (int ci = 0; ci < 4; ++ci) {
            aS[ci] = affA[ci]; aB[ci] = affA[4 + ci];
            bS[ci] = affB[ci]; bB[ci] = affB[4 + ci];
        }
#pragma unroll
        for (int co = 0; co < 4; ++co) bi[co] = bias[co];

        float m[4], s[4], q[4];
#pragma unroll
        for (int co = 0; co < 4; ++co) { m[co] = NEG_INF; s[co] = 0.f; q[co] = 0.f; }

        int r0 = 2 * oi - 1, c0 = 2 * oj - 1;
#pragma unroll
        for (int a0 = 0; a0 < 2; ++a0) {
            int r = r0 + a0;
            if (r < 0 || r >= IH) continue;
#pragma unroll
            for (int b0 = 0; b0 < 2; ++b0) {
                int c = c0 + b0;
                if (c < 0 || c >= IW) continue;
                float acc[4];
#pragma unroll
                for (int co = 0; co < 4; ++co) acc[co] = bi[co];
#pragma unroll
                for (int ci = 0; ci < 4; ++ci) {
                    const float* ipa = PA + ((long)na * 4 + ci) * IH * IW;
                    const float* ipb = PB + ((long)nb * 4 + ci) * IH * IW;
#pragma unroll
                    for (int di = 0; di < 3; ++di) {
                        int rr = r + di - 1;
                        if (rr < 0 || rr >= IH) continue;
#pragma unroll
                        for (int dj = 0; dj < 3; ++dj) {
                            int cc = c + dj - 1;
                            if (cc < 0 || cc >= IW) continue;
                            float va = fmaxf(0.f, aS[ci] * ipa[rr * IW + cc] + aB[ci]);
                            float vb = fmaxf(0.f, bS[ci] * ipb[rr * IW + cc] + bB[ci]);
                            float v = va - vb;
#pragma unroll
                            for (int co = 0; co < 4; ++co)
                                acc[co] += v * wgt[(co * 4 + ci) * 9 + di * 3 + dj];
                        }
                    }
                }
#pragma unroll
                for (int co = 0; co < 4; ++co) {
                    m[co] = fmaxf(m[co], acc[co]);
                    s[co] += acc[co];
                    q[co] += acc[co] * acc[co];
                }
            }
        }
#pragma unroll
        for (int co = 0; co < 4; ++co) {
            outP[((long)n * 4 + co) * (OH * OW) + rem] = m[co];
            part[co]     = s[co];
            part[4 + co] = q[co];
        }
    }
    block_reduce_store<8>(part, partials + (long)blockIdx.x * 8, lds);
}

// ---------------------------------------------------------------------------
// Head: per 16-edge tile (one wave32 each, 4 waves/block):
//   v64 = relu(aff6(P6)); feature = leaky0.1(fc(v))      -> 4x WMMA f32=f16xf16
//   h   = leaky0.01(reg1(feature))                       -> 1x WMMA
//   ypre = reg4(h)                                       -> LDS reduce
// WMMA fragment layouts per CDNA5 ISA 7.12.2 (16-bit A 16x32, col-per-lane B).
// ---------------------------------------------------------------------------
__global__ void head_kernel(const float* __restrict__ P6, const float* __restrict__ aff6,
                            const float* __restrict__ fcw, const float* __restrict__ fcb,
                            const float* __restrict__ r1w, const float* __restrict__ r1b,
                            const float* __restrict__ r4w, const float* __restrict__ r4b,
                            float* __restrict__ out, int E)
{
    __shared__ _Float16 sV[4][16][64];   // relu'd stage-6 vectors (f16)
    __shared__ _Float16 sF[4][16][32];   // feature (f16) for reg1 A-fragment
    __shared__ float    sH[4][16][16];   // h * r4w contributions

    const int w    = threadIdx.x >> 5;
    const int lane = threadIdx.x & 31;
    const int tile = blockIdx.x * 4 + w;
    const int e0   = tile * 16;

    // Phase 1: load + BN affine + relu -> f16 LDS
    for (int t = lane; t < 16 * 64; t += 32) {
        int e = t >> 6, k = t & 63, c = k >> 3;
        float raw = P6[(long)(e0 + e) * 64 + k];
        float v = fmaxf(0.f, aff6[c] * raw + aff6[8 + c]);
        sV[w][e][k] = (_Float16)v;
    }
    __syncthreads();

    const int M  = lane & 15;   // row (A) / column-N (B,D)
    const int hv = lane >> 4;   // lane half

    // Phase 2: fc GEMM, two N-tiles x two K-blocks of 32
#pragma unroll
    for (int nt = 0; nt < 2; ++nt) {
        v8f c8 = {};
#pragma unroll
        for (int t = 0; t < 2; ++t) {
            v16h A, B;
#pragma unroll
            for (int e = 0; e < 16; ++e) {
                int KA = t * 32 + ((e & 8) ? 16 : 0) + (hv ? 8 : 0) + (e & 7);
                A[e] = sV[w][M][KA];
                int KB = t * 32 + ((e & 8) ? 16 : 0) + (hv ? 8 : 0) + (e & 7);
                B[e] = (_Float16)fcw[(nt * 16 + M) * 64 + KB];
            }
            c8 = __builtin_amdgcn_wmma_f32_16x16x32_f16(false, A, false, B,
                                                        (short)0, c8, false, false);
        }
        int ncol = nt * 16 + M;
#pragma unroll
        for (int i = 0; i < 8; ++i) {
            int m = i + (hv ? 8 : 0);
            float f = c8[i] + fcb[ncol];
            f = f > 0.f ? f : 0.1f * f;                          // leaky_relu 0.1
            out[(long)E + (long)(e0 + m) * 32 + ncol] = f;       // feature output
            sF[w][m][ncol] = (_Float16)f;
        }
    }
    __syncthreads();

    // Phase 3: reg1 (32->16) via one WMMA, then h*r4w into LDS
    {
        v16h A, B;
#pragma unroll
        for (int e = 0; e < 16; ++e) {
            int KA = ((e & 8) ? 16 : 0) + (hv ? 8 : 0) + (e & 7);
            A[e] = sF[w][M][KA];
            int KB = ((e & 8) ? 16 : 0) + (hv ? 8 : 0) + (e & 7);
            B[e] = (_Float16)r1w[M * 32 + KB];
        }
        v8f c8 = {};
        c8 = __builtin_amdgcn_wmma_f32_16x16x32_f16(false, A, false, B,
                                                    (short)0, c8, false, false);
#pragma unroll
        for (int i = 0; i < 8; ++i) {
            int m = i + (hv ? 8 : 0);
            float h = c8[i] + r1b[M];
            h = h > 0.f ? h : 0.01f * h;                         // leaky_relu 0.01
            sH[w][m][M] = h * r4w[M];
        }
    }
    __syncthreads();

    // Phase 4: reg4 (16->1)
    if (lane < 16) {
        float s = r4b[0];
#pragma unroll
        for (int j = 0; j < 16; ++j) s += sH[w][lane][j];
        out[e0 + lane] = s;
    }
}

// ---------------------------------------------------------------------------
// Host driver
// ---------------------------------------------------------------------------
extern "C" void kernel_launch(void* const* d_in, const int* in_sizes, int n_in,
                              void* d_out, int out_size, void* d_ws, size_t ws_size,
                              hipStream_t stream)
{
    (void)out_size; (void)ws_size;

    const float* x  = (const float*)d_in[0];
    const int*   ei = (const int*)d_in[1];

    // conv params, flattened in setup_inputs() dict order: w,b,g,beta per block
    const float *w_c1  = (const float*)d_in[2],  *b_c1  = (const float*)d_in[3],
                *g_c1  = (const float*)d_in[4],  *be_c1 = (const float*)d_in[5];
    const float *w_c1b = (const float*)d_in[6],  *b_c1b = (const float*)d_in[7],
                *g_c1b = (const float*)d_in[8],  *be_c1b= (const float*)d_in[9];
    const float *w_c2  = (const float*)d_in[10], *b_c2  = (const float*)d_in[11],
                *g_c2  = (const float*)d_in[12], *be_c2 = (const float*)d_in[13];
    const float *w_c2b = (const float*)d_in[14], *b_c2b = (const float*)d_in[15],
                *g_c2b = (const float*)d_in[16], *be_c2b= (const float*)d_in[17];
    const float *w_c3  = (const float*)d_in[18], *b_c3  = (const float*)d_in[19],
                *g_c3  = (const float*)d_in[20], *be_c3 = (const float*)d_in[21];
    const float *w_c4  = (const float*)d_in[22], *b_c4  = (const float*)d_in[23],
                *g_c4  = (const float*)d_in[24], *be_c4 = (const float*)d_in[25];
    const float *w_c5  = (const float*)d_in[26], *b_c5  = (const float*)d_in[27],
                *g_c5  = (const float*)d_in[28], *be_c5 = (const float*)d_in[29];
    const float *w_c6  = (const float*)d_in[30], *b_c6  = (const float*)d_in[31],
                *g_c6  = (const float*)d_in[32], *be_c6 = (const float*)d_in[33];

    // SE blocks contain zero-element arrays (hidden dim 0); walk skipping them
    // so we are robust to the harness either passing or dropping empties.
    int cur = 34;
    auto take = [&]() -> const float* {
        while (cur < n_in && in_sizes[cur] == 0) ++cur;
        return (const float*)d_in[cur++];
    };
    const float* se2b2 = take();          // [4]
    const float* se3b2 = take();          // [8]
    const float* fcw = take(); const float* fcb = take();   // [32,64],[32]
    const float* r1w = take(); const float* r1b = take();   // [16,32],[16]
    const float* r4w = take(); const float* r4b = take();   // [16],[1]

    // ---- workspace layout ----
    char* base = (char*)d_ws;
    size_t off = 0;
    auto alloc = [&](size_t bytes) -> void* {
        void* p = base + off;
        off += (bytes + 255) & ~(size_t)255;
        return p;
    };
    int* cnt0 = (int*)alloc(N_NODES * sizeof(int));
    int* cnt1 = (int*)alloc(N_NODES * sizeof(int));
    float* aff1a = (float*)alloc(8 * sizeof(float));
    float* aff1b = (float*)alloc(8 * sizeof(float));
    float* aff2a = (float*)alloc(8 * sizeof(float));
    float* aff2b = (float*)alloc(8 * sizeof(float));
    float* aff3  = (float*)alloc(8 * sizeof(float));
    float* aff4  = (float*)alloc(16 * sizeof(float));
    float* aff5  = (float*)alloc(16 * sizeof(float));
    float* aff6  = (float*)alloc(16 * sizeof(float));

    float* P1a = (float*)alloc((size_t)N_NODES * 4 * 161 * 51 * 4);
    float* P1b = (float*)alloc((size_t)N_NODES * 4 * 161 * 51 * 4);
    float* P2a = (float*)alloc((size_t)N_NODES * 4 * 81 * 26 * 4);
    float* P2b = (float*)alloc((size_t)N_NODES * 4 * 81 * 26 * 4);
    float* P3  = (float*)alloc((size_t)E_EDGES * 4 * 41 * 14 * 4);
    float* P4  = (float*)alloc((size_t)E_EDGES * 8 * 21 * 8 * 4);
    float* P5  = (float*)alloc((size_t)E_EDGES * 8 * 11 * 5 * 4);
    float* P6  = (float*)alloc((size_t)E_EDGES * 64 * 4);

    const int g1 = (N_NODES * 161 * 51 + 255) / 256;   // 16422
    const int g2 = (N_NODES * 81 * 26 + 255) / 256;    // 4212
    const int g3 = (E_EDGES * 41 * 14 + 255) / 256;    // 4592
    const int g4 = (E_EDGES * 21 * 8 + 255) / 256;     // 1344
    const int g5 = (E_EDGES * 11 * 5 + 255) / 256;     // 440
    const int g6 = (E_EDGES * 4 * 2 + 255) / 256;      // 64

    float* part1  = (float*)alloc((size_t)g1 * 16 * 4);
    float* part2a = (float*)alloc((size_t)g2 * 8 * 4);
    float* part2b = (float*)alloc((size_t)g2 * 8 * 4);
    float* part3  = (float*)alloc((size_t)g3 * 8 * 4);
    float* part4  = (float*)alloc((size_t)g4 * 16 * 4);
    float* part5  = (float*)alloc((size_t)g5 * 16 * 4);
    float* part6  = (float*)alloc((size_t)g6 * 16 * 4);

    const float invT1 = 1.f / (2048.f * 320.f * 100.f);
    const float invT2 = 1.f / (2048.f * 161.f * 51.f);
    const float invT3 = 1.f / (2048.f * 81.f * 26.f);
    const float invT4 = 1.f / (2048.f * 41.f * 14.f);
    const float invT5 = 1.f / (2048.f * 21.f * 8.f);
    const float invT6 = 1.f / (2048.f * 11.f * 5.f);

    // ---- pipeline ----
    hipMemsetAsync(d_ws, 0, 4096, stream);   // zero cnt0/cnt1
    count_kernel<<<(E_EDGES + 255) / 256, 256, 0, stream>>>(ei, cnt0, cnt1, E_EDGES);

    conv1_kernel<<<g1, 256, 0, stream>>>(x, cnt0, cnt1, w_c1, b_c1, w_c1b, b_c1b,
                                         P1a, P1b, part1);
    bn_affine_kernel<<<1, 4, 0, stream>>>(part1, g1, 16, 0, 4, 4, g_c1,  be_c1,  invT1, aff1a);
    bn_affine_kernel<<<1, 4, 0, stream>>>(part1, g1, 16, 8, 12, 4, g_c1b, be_c1b, invT1, aff1b);

    conv_stage_kernel<4, 4, 2, 2><<<g2, 256, 0, stream>>>(P1a, aff1a, nullptr, w_c2, b_c2,
        cnt0, P2a, part2a, N_NODES, 161, 51, 81, 26);
    conv_stage_kernel<4, 4, 2, 2><<<g2, 256, 0, stream>>>(P1b, aff1b, nullptr, w_c2b, b_c2b,
        cnt1, P2b, part2b, N_NODES, 161, 51, 81, 26);
    bn_affine_kernel<<<1, 4, 0, stream>>>(part2a, g2, 8, 0, 4, 4, g_c2,  be_c2,  invT2, aff2a);
    bn_affine_kernel<<<1, 4, 0, stream>>>(part2b, g2, 8, 0, 4, 4, g_c2b, be_c2b, invT2, aff2b);

    conv3_diff_kernel<<<g3, 256, 0, stream>>>(P2a, P2b, aff2a, aff2b, ei, ei + E_EDGES,
        w_c3, b_c3, P3, part3, E_EDGES, 81, 26, 41, 14);
    bn_affine_kernel<<<1, 4, 0, stream>>>(part3, g3, 8, 0, 4, 4, g_c3, be_c3, invT3, aff3);

    conv_stage_kernel<4, 8, 2, 2><<<g4, 256, 0, stream>>>(P3, aff3, se2b2, w_c4, b_c4,
        nullptr, P4, part4, E_EDGES, 41, 14, 21, 8);
    bn_affine_kernel<<<1, 8, 0, stream>>>(part4, g4, 16, 0, 8, 8, g_c4, be_c4, invT4, aff4);

    conv_stage_kernel<8, 8, 2, 2><<<g5, 256, 0, stream>>>(P4, aff4, nullptr, w_c5, b_c5,
        nullptr, P5, part5, E_EDGES, 21, 8, 11, 5);
    bn_affine_kernel<<<1, 8, 0, stream>>>(part5, g5, 16, 0, 8, 8, g_c5, be_c5, invT5, aff5);

    conv_stage_kernel<8, 8, 3, 3><<<g6, 256, 0, stream>>>(P5, aff5, se3b2, w_c6, b_c6,
        nullptr, P6, part6, E_EDGES, 11, 5, 4, 2);
    bn_affine_kernel<<<1, 8, 0, stream>>>(part6, g6, 16, 0, 8, 8, g_c6, be_c6, invT6, aff6);

    head_kernel<<<E_EDGES / 64, 128, 0, stream>>>(P6, aff6, fcw, fcb, r1w, r1b, r4w, r4b,
                                                  (float*)d_out, E_EDGES);
}
